// AGCN_Encoder_62758062129202
// MI455X (gfx1250) — matrix operand order repaired and checked by reading, not compiled, not run on hardware
//
#include <hip/hip_runtime.h>
#include <stdint.h>

// ---------------- types ----------------
typedef __bf16 bf16;
typedef __bf16 v16bf __attribute__((ext_vector_type(16)));
typedef float  v8f   __attribute__((ext_vector_type(8)));
typedef unsigned int u32x4 __attribute__((ext_vector_type(4)));
typedef int          i32x4 __attribute__((ext_vector_type(4)));
typedef int          i32x8 __attribute__((ext_vector_type(8)));

#define B_   8
#define T_   24
#define N_   400
#define H_   64
#define EMB_ 32
#define M_   20
#define BT_  (B_*T_)
#define KCH  384            // 2*CHEB*AOD
#define NTI  25             // N/16
#define SP   416            // N padded to multiple of 32 (13 K-tiles, no tail logic)
#define SP2  (SP*SP)

#if defined(__has_builtin)
#  if __has_builtin(__builtin_amdgcn_tensor_load_to_lds) && __has_builtin(__builtin_amdgcn_s_wait_tensorcnt)
#    define CDNA5_TDM 1
#  endif
#endif
#ifndef CDNA5_TDM
#define CDNA5_TDM 0
#endif

// ---------------- WMMA helpers ----------------
__device__ __forceinline__ v8f wmma_bf16(v16bf a, v16bf b, v8f c) {
  return __builtin_amdgcn_wmma_f32_16x16x32_bf16(false, a, false, b, (short)0, c, false, false);
}

union frag_cvt { uint4 u[2]; v16bf v; };

// A fragment (16x32 tile, row-major, ld multiple of 8, base 16B aligned).
// ISA layout <=> memory: lane's 16 values = row[kq*8 .. +7] ++ row[16+kq*8 .. +7]
// -> exactly two b128 loads per lane.
template <typename PTR>
__device__ __forceinline__ v16bf load_a_frag(PTR base, int ld) {
  int l = threadIdx.x & 31;
  int row = l & 15, kq = l >> 4;
  const bf16* p = base + row * ld + kq * 8;
  frag_cvt c;
  c.u[0] = *(const uint4*)p;
  c.u[1] = *(const uint4*)(p + 16);
  return c.v;
}

// B fragment (32x16 tile) from a COLUMN-MAJOR source: element (k, col) at baseT[col*ldT + k].
// Lane's 16 values = column[kq*8 .. +7] ++ column[16+kq*8 .. +7] -> two b128 loads per lane.
template <typename PTR>
__device__ __forceinline__ v16bf load_b_fragT(PTR baseT, int ldT) {
  int l = threadIdx.x & 31;
  int col = l & 15, kq = l >> 4;
  const bf16* p = baseT + col * ldT + kq * 8;
  frag_cvt c;
  c.u[0] = *(const uint4*)p;
  c.u[1] = *(const uint4*)(p + 16);
  return c.v;
}

__device__ __forceinline__ float wave_sum(float v) {
#pragma unroll
  for (int o = 16; o > 0; o >>= 1) v += __shfl_xor(v, o, 32);
  return v;
}

// ---------------- K1: zero-padded S stack (I, s) + transposed s + transposed gcn_w ----------------
__global__ void k_prep(const float* __restrict__ supports, const float* __restrict__ gcn_w,
                       bf16* __restrict__ S_bf, bf16* __restrict__ S_T,
                       bf16* __restrict__ gcnwT, float* __restrict__ loss) {
  int i = blockIdx.x * blockDim.x + threadIdx.x;
  if (i < 8) loss[i] = 0.f;
  if (i < SP2) {
    int row = i / SP, col = i % SP;
    bf16 z = (bf16)0.0f;
#pragma unroll
    for (int m = 0; m < 6; ++m) S_bf[m * SP2 + i] = z;
    S_T[0 * SP2 + i] = z;
    S_T[1 * SP2 + i] = z;
    if (row < N_ && col < N_) {
      bf16 eye = (bf16)((row == col) ? 1.f : 0.f);
      S_bf[0 * SP2 + i] = eye;
      S_bf[3 * SP2 + i] = eye;
      S_bf[1 * SP2 + i] = (bf16)supports[row * N_ + col];
      S_bf[4 * SP2 + i] = (bf16)supports[N_ * N_ + row * N_ + col];
      // transposed copies (treat i as [col-major] slot: S_T[c][r])
      S_T[0 * SP2 + i] = (bf16)supports[col * N_ + row];
      S_T[1 * SP2 + i] = (bf16)supports[N_ * N_ + col * N_ + row];
    }
  }
  if (i < KCH * H_) {
    int k = i >> 6, h = i & 63;
    gcnwT[h * KCH + k] = (bf16)gcn_w[i];
  }
}

// ---------------- K2: S[3s+2] = 2*s@s - I via bf16 WMMA (padded K=416, no tails) ----------------
__global__ void __launch_bounds__(32)
k_cheb2(bf16* __restrict__ S_bf, const bf16* __restrict__ S_T) {
  int rb = blockIdx.x, cb = blockIdx.y, s = blockIdx.z;
  const bf16* A = S_bf + (3 * s + 1) * SP2 + rb * 16 * SP;
  const bf16* BT = S_T + s * SP2 + cb * 16 * SP;
  v8f acc = {};
  for (int k0 = 0; k0 < SP; k0 += 32) {
    v16bf a = load_a_frag(A + k0, SP);
    v16bf b = load_b_fragT(BT + k0, SP);
    acc = wmma_bf16(a, b, acc);
  }
  int l = threadIdx.x & 31;
  int colg = cb * 16 + (l & 15);
  bf16* D = S_bf + (3 * s + 2) * SP2;
#pragma unroll
  for (int r = 0; r < 8; ++r) {
    int rowg = rb * 16 + ((l >> 4) * 8) + r;
    float v = 2.f * acc[r] - ((rowg == colg) ? 1.f : 0.f);
    D[rowg * SP + colg] = (bf16)v;
  }
}

// ---------------- K3: SDNE encode (store emb) + fused decode/loss, one x_od pass ----------------
__global__ void __launch_bounds__(128)
k_sdne(const float* __restrict__ x_od, const float* __restrict__ We, const float* __restrict__ be,
       const float* __restrict__ Wd, const float* __restrict__ bd,
       float* __restrict__ emb, float* __restrict__ loss) {
  __shared__ float sRow[4][N_];
  __shared__ float sEmb[4][EMB_];
  int w = threadIdx.x >> 5, lane = threadIdx.x & 31;
  int rid = blockIdx.x * 4 + w;                    // row in [0, B*T*N)
  const float* row = x_od + (size_t)rid * N_;
  float deg = 0.f;
  for (int m = lane; m < N_; m += 32) { float v = row[m]; sRow[w][m] = v; deg += v; }
  deg = wave_sum(deg);
  float e_acc = be[lane];                          // lane == embedding channel (EMB==32)
  for (int m = 0; m < N_; ++m) e_acc += sRow[w][m] * We[m * EMB_ + lane];
  e_acc = fmaxf(e_acc, 0.f);
  sEmb[w][lane] = e_acc;
  emb[(size_t)rid * EMB_ + lane] = e_acc;
  float z2 = wave_sum(e_acc * e_acc);
  float sq = 0.f;
  for (int m = lane; m < N_; m += 32) {
    float hsum = bd[m];
#pragma unroll
    for (int e = 0; e < EMB_; ++e) hsum += sEmb[w][e] * Wd[e * N_ + m];
    hsum = fmaxf(hsum, 0.f);
    float xv = sRow[w][m];
    float beta = (xv != 0.f) ? 5.f : 1.f;
    float d = (xv - hsum) * beta;
    sq += d * d;
  }
  sq = wave_sum(sq);
  if (lane == 0) {
    float c = (sq * (1.f / (B_ * (float)N_)) + (0.04f / B_) * deg * z2) * (1.f / T_);
    atomicAdd(loss, c);
  }
}

// ---------------- K4: temporal MHA (causal, per node) + FF + layernorm -> bf16 (TRANSPOSED) ----------------
__global__ void __launch_bounds__(64)
k_attention(const float* __restrict__ x,
            const float* __restrict__ Wq, const float* __restrict__ bq,
            const float* __restrict__ Wk, const float* __restrict__ bk,
            const float* __restrict__ Wv, const float* __restrict__ bv,
            const float* __restrict__ Wtx, const float* __restrict__ btx,
            const float* __restrict__ Wff1, const float* __restrict__ bff1,
            const float* __restrict__ Wff2, const float* __restrict__ bff2,
            const float* __restrict__ ln_g, const float* __restrict__ ln_b,
            bf16* __restrict__ attnT) {          // [bt][64][400] column-major per (b,t)
  __shared__ float sX[T_ * 2];
  __shared__ float sQ[T_*H_], sK[T_*H_], sV[T_*H_], sO[T_*H_], sG[T_*H_];
  int n = blockIdx.x, b = blockIdx.y;
  int h = threadIdx.x;
  if (h < T_ * 2) {
    int t = h >> 1, c = h & 1;
    sX[h] = x[(((b * T_ + t) * N_) + n) * 2 + c];
  }
  __syncthreads();
  for (int t = 0; t < T_; ++t) {
    float x0 = sX[2*t], x1 = sX[2*t + 1];
    sQ[t*H_ + h] = x0*Wq[h] + x1*Wq[H_ + h] + bq[h];
    sK[t*H_ + h] = x0*Wk[h] + x1*Wk[H_ + h] + bk[h];
    sV[t*H_ + h] = x0*Wv[h] + x1*Wv[H_ + h] + bv[h];
  }
  __syncthreads();
  for (int qt = 0; qt < T_; ++qt) {              // online-softmax causal attention
    float m = -1e30f, l = 0.f, acc = 0.f;
    float qv = sQ[qt*H_ + h];
    for (int kk = 0; kk <= qt; ++kk) {
      float p = qv * sK[kk*H_ + h];
      p += __shfl_xor(p, 1, 16);
      p += __shfl_xor(p, 2, 16);
      p += __shfl_xor(p, 4, 16);
      p += __shfl_xor(p, 8, 16);
      float s = p * 0.25f;                       // 1/sqrt(d=16)
      float mn = fmaxf(m, s);
      float cold = __expf(m - mn), e = __expf(s - mn);
      l = l*cold + e;
      acc = acc*cold + e * sV[kk*H_ + h];
      m = mn;
    }
    sO[qt*H_ + h] = acc / l;
  }
  __syncthreads();
  for (int t = 0; t < T_; ++t) {                 // FF1 + exact gelu
    float f = bff1[h];
    for (int j = 0; j < H_; ++j) f += sO[t*H_ + j] * Wff1[j*H_ + h];
    sG[t*H_ + h] = 0.5f * f * (1.f + erff(f * 0.70710678118654752f));
  }
  __syncthreads();
  for (int t = 0; t < T_; ++t) {                 // FF2 + x@Wtx path (reuse sK)
    float f = bff2[h];
    for (int j = 0; j < H_; ++j) f += sG[t*H_ + j] * Wff2[j*H_ + h];
    float x0 = sX[2*t], x1 = sX[2*t + 1];
    sK[t*H_ + h] = f + x0*Wtx[h] + x1*Wtx[H_ + h] + btx[h];
  }
  __syncthreads();
  for (int t = 0; t < T_; ++t) {                 // layernorm, write bf16 column-major
    float su = 0.f, sqv = 0.f;
    for (int j = 0; j < H_; ++j) { float v = sK[t*H_ + j]; su += v; sqv += v*v; }
    float mu = su * (1.f / H_);
    float var = sqv * (1.f / H_) - mu * mu;
    float v = (sK[t*H_ + h] - mu) * rsqrtf(var + 1e-5f) * ln_g[h] + ln_b[h];
    attnT[(((size_t)(b * T_ + t)) * H_ + h) * N_ + n] = (bf16)v;
  }
}

// ---------------- K5: fused Chebyshev diffusion (6 supports) + GCN, bf16 WMMA ----------------
__global__ void __launch_bounds__(128)
k_diff_gcn(const bf16* __restrict__ S_bf, const bf16* __restrict__ attnT,
           const bf16* __restrict__ gcnwT, const float* __restrict__ gcn_b,
           float* __restrict__ out) {
  __shared__ __align__(16) bf16 sAttnT[H_ * N_ + 32];  // column-major [64][400] + tail slack
  __shared__ __align__(16) bf16 sXg[16 * KCH];         // xg tile (stage1 out / stage2 A)
  int nb = blockIdx.x, bt = blockIdx.y;
  const bf16* g = attnT + (size_t)bt * H_ * N_;
#if CDNA5_TDM
  if (threadIdx.x < 32) {                        // wave 0: TDM flat 25600-elem tile load
    unsigned int ldsaddr = (unsigned int)(uintptr_t)(&sAttnT[0]);
    unsigned long long ga = (unsigned long long)(uintptr_t)g;
    u32x4 g0;
    g0[0] = 1u;                                  // count=1
    g0[1] = ldsaddr;                             // lds_addr
    g0[2] = (unsigned int)ga;                    // global_addr[31:0]
    g0[3] = (unsigned int)(ga >> 32) | (2u << 30); // global_addr[56:32] | type=2
    i32x8 g1;
    g1[0] = (int)(1u << 16);                     // data_size = 2B
    g1[1] = (int)((unsigned)(H_ * N_) << 16);    // tensor_dim0 = 25600 (bits 79:48 low16)
    g1[2] = (int)(1u << 16) | (int)((unsigned)(H_ * N_) >> 16); // dim0 hi | tensor_dim1=1
    g1[3] = (int)((unsigned)(H_ * N_) << 16);    // tile_dim0 = 25600 (bits 127:112)
    g1[4] = 1;                                   // tile_dim1 = 1, tile_dim2 = 0
    g1[5] = (int)(unsigned)(H_ * N_);            // tensor_dim0_stride
    g1[6] = 0; g1[7] = 0;
    i32x4 gz = {0, 0, 0, 0};
#if defined(__clang_major__) && (__clang_major__ >= 23)
    i32x8 gz8 = {0, 0, 0, 0, 0, 0, 0, 0};
    __builtin_amdgcn_tensor_load_to_lds(g0, g1, gz, gz, gz8, 0);
#else
    __builtin_amdgcn_tensor_load_to_lds(g0, g1, gz, gz, 0);
#endif
    __builtin_amdgcn_s_wait_tensorcnt(0);
  }
#else
  {
    const uint4* src = (const uint4*)g;
    uint4* dst = (uint4*)sAttnT;
    for (int i = threadIdx.x; i < (H_ * N_) / 8; i += blockDim.x) dst[i] = src[i];
  }
#endif
  if (threadIdx.x >= 32 && threadIdx.x < 64)     // zero tail slack (wave 1)
    sAttnT[H_ * N_ + (threadIdx.x - 32)] = (bf16)0.0f;
  __syncthreads();

  int w = threadIdx.x >> 5;
  int c0 = w * 16;
  int lane = threadIdx.x & 31;
  int colr = lane & 15, rb8 = (lane >> 4) * 8;
  // stage 1: xg[:, s*64 + c] = S_s @ attn ; each wave owns a 16-col slab.
  // K padded to 416: S cols 400..415 are zero, so tail B reads (garbage-but-finite) are annulled.
#pragma unroll 1
  for (int s = 0; s < 6; ++s) {
    v8f acc = {};
    const bf16* Sb = S_bf + s * SP2 + nb * 16 * SP;
    const bf16* Bt = sAttnT + c0 * N_;
#pragma unroll 1
    for (int k0 = 0; k0 < SP; k0 += 32) {
      if (k0 + 32 < SP) __builtin_prefetch((const void*)(Sb + k0 + 32), 0, 3);
      v16bf a = load_a_frag(Sb + k0, SP);
      v16bf b = load_b_fragT(Bt + k0, N_);
      acc = wmma_bf16(a, b, acc);
    }
#pragma unroll
    for (int r = 0; r < 8; ++r)
      sXg[(rb8 + r) * KCH + s * H_ + c0 + colr] = (bf16)acc[r];
  }
  __syncthreads();
  // stage 2: state = xg[16,384] @ gcn_w[384,64] (A from LDS, B pre-transposed)
  v8f acc = {};
#pragma unroll 1
  for (int k0 = 0; k0 < KCH; k0 += 32) {
    v16bf a = load_a_frag(sXg + k0, KCH);
    v16bf b = load_b_fragT(gcnwT + c0 * KCH + k0, KCH);
    acc = wmma_bf16(a, b, acc);
  }
  float* o = out + ((size_t)bt * N_ + nb * 16) * H_;
  int col = c0 + colr;
  float bias = gcn_b[col];
#pragma unroll
  for (int r = 0; r < 8; ++r)
    o[(rb8 + r) * H_ + col] = acc[r] + bias;
}

// ---------------- K6: odw -> xod_g -> query -> memory attention, value + triplet loss ----------------
__global__ void __launch_bounds__(256)
k_mem(const float* __restrict__ out, const float* __restrict__ emb,
      const float* __restrict__ Wod, const float* __restrict__ bod,
      const float* __restrict__ Wq_mem, const float* __restrict__ Memory,
      float* __restrict__ value_ws, float* __restrict__ loss) {
  __shared__ float sOdw[N_];
  __shared__ float sE[EMB_];
  __shared__ float sQm[H_];
  __shared__ float sAtt[M_];
  __shared__ int sIdx[2];
  int bt = blockIdx.x, tid = threadIdx.x;
  const float* state = out + (size_t)bt * N_ * H_;
  for (int n = tid; n < N_; n += 256) {
    float a = bod[0];
    for (int h = 0; h < H_; ++h) a += state[n * H_ + h] * Wod[h];
    sOdw[n] = tanhf(a);
  }
  __syncthreads();
  if (tid < EMB_) {
    float a = 0.f;
    for (int n = 0; n < N_; ++n) a += sOdw[n] * emb[((size_t)bt * N_ + n) * EMB_ + tid];
    sE[tid] = a;
  }
  __syncthreads();
  if (tid < H_) {
    float a = 0.f;
#pragma unroll
    for (int e = 0; e < EMB_; ++e) a += sE[e] * Wq_mem[e * H_ + tid];
    sQm[tid] = a;
  }
  __syncthreads();
  if (tid < M_) {
    float a = 0.f;
    for (int h = 0; h < H_; ++h) a += sQm[h] * Memory[tid * H_ + h];
    sAtt[tid] = a;
  }
  __syncthreads();
  if (tid == 0) {
    float mx = -1e30f;
    for (int m = 0; m < M_; ++m) mx = fmaxf(mx, sAtt[m]);
    float s = 0.f;
    for (int m = 0; m < M_; ++m) { sAtt[m] = __expf(sAtt[m] - mx); s += sAtt[m]; }
    float inv = 1.f / s;
    for (int m = 0; m < M_; ++m) sAtt[m] *= inv;
    int i0 = 0;
    for (int m = 1; m < M_; ++m) if (sAtt[m] > sAtt[i0]) i0 = m;
    int i1 = (i0 == 0) ? 1 : 0;
    for (int m = 0; m < M_; ++m) if (m != i0 && sAtt[m] > sAtt[i1]) i1 = m;
    sIdx[0] = i0; sIdx[1] = i1;
  }
  __syncthreads();
  if (tid < H_) {
    float v = 0.f;
    for (int m = 0; m < M_; ++m) v += sAtt[m] * Memory[m * H_ + tid];
    value_ws[bt * H_ + tid] = v;
  }
  if (tid == 0) {
    int i0 = sIdx[0], i1 = sIdx[1];
    float dp = 0.f, dn = 0.f, comp = 0.f;
    for (int h = 0; h < H_; ++h) {
      float dq = sQm[h] - Memory[i0 * H_ + h];
      float a = dq + 1e-6f;  dp += a * a;  comp += dq * dq;
      float b = sQm[h] - Memory[i1 * H_ + h] + 1e-6f;  dn += b * b;
    }
    dp = sqrtf(dp); dn = sqrtf(dn);
    float trip = fmaxf(0.f, dp - dn + 1.f) * (1.f / (float)BT_);
    float cmp = comp * (1.f / ((float)BT_ * (float)H_));
    atomicAdd(loss, trip + cmp);
  }
}

// ---------------- K7: out += value broadcast, emit scalar loss ----------------
__global__ void k_finalize(float* __restrict__ out, const float* __restrict__ value_ws,
                           const float* __restrict__ loss) {
  int i = blockIdx.x * blockDim.x + threadIdx.x;
  if (i < BT_ * N_ * H_) {
    int bt = i / (N_ * H_);
    int h = i & (H_ - 1);
    out[i] += value_ws[bt * H_ + h];
  }
  if (i == 0) out[BT_ * N_ * H_] = loss[0];
}

// ---------------- launch ----------------
extern "C" void kernel_launch(void* const* d_in, const int* in_sizes, int n_in,
                              void* d_out, int out_size, void* d_ws, size_t ws_size,
                              hipStream_t stream) {
  (void)in_sizes; (void)n_in; (void)out_size; (void)ws_size;
  const float* x        = (const float*)d_in[0];
  const float* x_od     = (const float*)d_in[1];
  const float* supports = (const float*)d_in[2];
  const float* Wq   = (const float*)d_in[3];  const float* bq   = (const float*)d_in[4];
  const float* Wk   = (const float*)d_in[5];  const float* bk   = (const float*)d_in[6];
  const float* Wv   = (const float*)d_in[7];  const float* bv   = (const float*)d_in[8];
  const float* Wtx  = (const float*)d_in[9];  const float* btx  = (const float*)d_in[10];
  const float* Wff1 = (const float*)d_in[11]; const float* bff1 = (const float*)d_in[12];
  const float* Wff2 = (const float*)d_in[13]; const float* bff2 = (const float*)d_in[14];
  const float* ln_g = (const float*)d_in[15]; const float* ln_b = (const float*)d_in[16];
  const float* gcn_w = (const float*)d_in[17]; const float* gcn_b = (const float*)d_in[18];
  const float* Wod  = (const float*)d_in[19]; const float* bod  = (const float*)d_in[20];
  const float* We   = (const float*)d_in[21]; const float* be   = (const float*)d_in[22];
  const float* Wd   = (const float*)d_in[23]; const float* bd   = (const float*)d_in[24];
  const float* Wq_mem = (const float*)d_in[25]; const float* Memory = (const float*)d_in[26];
  float* out = (float*)d_out;

  // workspace carve-up (16B aligned, ~25 MB total)
  char* ws = (char*)d_ws;
  float* loss    = (float*)(ws + 0);                  // 8 f32
  float* value_w = (float*)(ws + 1024);               // BT*64 f32
  float* emb     = (float*)(ws + (1u  << 20));        // 9.83 MB f32
  bf16* S_bf     = (bf16*)(ws + (11u << 20));         // 6*416*416 bf16 ~ 2.08 MB (zero-padded)
  bf16* S_T      = (bf16*)(ws + (14u << 20));         // 2*416*416 bf16 (transposed supports)
  bf16* gcnwT    = (bf16*)(ws + (15u << 20));         // 64*384 bf16 (transposed gcn_w)
  bf16* attnT    = (bf16*)(ws + (16u << 20));         // 9.83 MB bf16, [bt][64][400]

  k_prep<<<dim3((SP2 + 255) / 256), 256, 0, stream>>>(supports, gcn_w, S_bf, S_T, gcnwT, loss);
  k_cheb2<<<dim3(NTI, NTI, 2), 32, 0, stream>>>(S_bf, S_T);
  k_sdne<<<dim3(BT_ * N_ / 4), 128, 0, stream>>>(x_od, We, be, Wd, bd, emb, loss);
  k_attention<<<dim3(N_, B_), 64, 0, stream>>>(x, Wq, bq, Wk, bk, Wv, bv, Wtx, btx,
                                               Wff1, bff1, Wff2, bff2, ln_g, ln_b, attnT);
  k_diff_gcn<<<dim3(NTI, BT_), 128, 0, stream>>>(S_bf, attnT, gcnwT, gcn_b, out);
  k_mem<<<dim3(BT_), 256, 0, stream>>>(out, emb, Wod, bod, Wq_mem, Memory, value_w, loss);
  k_finalize<<<dim3((BT_ * N_ * H_ + 255) / 256), 256, 0, stream>>>(out, value_w, loss);
}
